// TransformerEncoder_35356170780986
// MI455X (gfx1250) — compile-verified, matrix operand order checked by reading, not run on hardware
//
#include <hip/hip_runtime.h>
#include <hip/hip_bf16.h>

// ---- problem constants (from reference) ----
#define L_ 3
#define B_ 2
#define S_ 2048
#define D_ 256
#define H_ 8
#define F_ 1024
#define DK_ 32
#define NEG_INF (-3.402823466e38f)

typedef __attribute__((ext_vector_type(16))) __bf16 v16bf;
typedef __attribute__((ext_vector_type(8)))  __bf16 v8bf;
typedef __attribute__((ext_vector_type(8)))  float  v8f;

// Fragment K-runs (ISA 7.12.2): per-lane data = two contiguous 16B runs at
// K = g*8 and K = 16+g*8 (g = lane>>4), striped by r = lane&15.

__device__ __forceinline__ v8f wmma_bf16(v16bf a, v16bf b, v8f c) {
  return __builtin_amdgcn_wmma_f32_16x16x32_bf16(false, a, false, b, (short)0, c,
                                                 false, false);
}

__device__ __forceinline__ v16bf cat16(v8bf lo, v8bf hi) {
  union { v16bf v; v8bf h[2]; } u;
  u.h[0] = lo; u.h[1] = hi;
  return u.v;
}

// CDNA5 async copy: global -> LDS, 16B per lane, tracked by ASYNCcnt.
// Generic LDS pointer's low 32 bits are the LDS byte address (flat aperture rule).
__device__ __forceinline__ void async_b128(void* lds_ptr, const void* gptr) {
  unsigned lds_off = (unsigned)(unsigned long long)lds_ptr;
  asm volatile("global_load_async_to_lds_b128 %0, %1, off"
               :: "v"(lds_off), "v"(gptr)
               : "memory");
}
__device__ __forceinline__ void wait_async() {
  asm volatile("s_wait_asynccnt 0" ::: "memory");
}
// Wave-private LDS store->load ordering fence (DS ops are in issue order per wave;
// this only stops compiler reordering and forces the dependency wait).
__device__ __forceinline__ void lds_fence() {
  asm volatile("s_wait_dscnt 0" ::: "memory");
}

// ---------------- weight convert + transpose (fp32 [R][C] -> bf16 [C][R]) ----------------
__global__ __launch_bounds__(256) void transpose_bf16_kernel(
    const float* __restrict__ in, __bf16* __restrict__ out, int R, int C) {
  long long base = (long long)blockIdx.y * R * C;
  int i = blockIdx.x * 256 + threadIdx.x;
  if (i < R * C) {
    int rr = i / C, cc = i % C;
    out[base + (long long)cc * R + rr] = (__bf16)in[base + i];
  }
}

// ---------------- x = x + pe (fp32 + bf16 outputs) ----------------
__global__ __launch_bounds__(256) void addpe_kernel(const float* __restrict__ x,
                                                    const float* __restrict__ pe,
                                                    float* __restrict__ outf,
                                                    __bf16* __restrict__ outb) {
  long long i = (long long)blockIdx.x * blockDim.x + threadIdx.x;
  int sd = (int)(i % (long long)(S_ * D_));
  float vv = x[i] + pe[sd];
  outf[i] = vv;
  outb[i] = (__bf16)vv;
}

// ---------------- bf16-WMMA GEMM, async-staged, double-buffered ----------------
// Bt is pre-transposed [N][K] (ldbt = row stride in elems).
// EPI 0: Cb bf16 with (batch,row,col)-stride mapping (QKV projections)
// EPI 1: Cf fp32 plain [M,N], + bias + residual          (out-proj / FFN2)
// EPI 2: Cb bf16 plain [M,N], + bias + relu              (FFN1)
// Block = 256 threads (8 waves), macro-tile 64(M) x 32(N), K-step 64.
template <int EPI>
__global__ __launch_bounds__(256) void gemm_wmma_kernel(
    const __bf16* __restrict__ A, int lda, const __bf16* __restrict__ Bt, int ldbt,
    const float* __restrict__ bias, const float* __restrict__ residual,
    float* __restrict__ Cf, __bf16* __restrict__ Cb,
    int rowsPerBatch, long long batchStrideC, int rowStrideC, long long colStrideC,
    int N, int K) {
  __shared__ __bf16 As[2][64][72];  // [buf][m][k], 144B rows
  __shared__ __bf16 Bs[2][32][72];  // [buf][n][k]
  const int t = threadIdx.x;
  const int lane = t & 31;
  const int w = t >> 5;
  const int g = lane >> 4;
  const int r = lane & 15;
  const int m0 = blockIdx.y * 64;
  const int n0 = blockIdx.x * 32;
  const int rowTile = w >> 1;  // 0..3
  const int colTile = w & 1;   // 0..1
  const int ar = rowTile * 16 + r;
  const int bn = colTile * 16 + r;

  auto stageAB = [&](int buf, int k0) {
#pragma unroll
    for (int j = 0; j < 2; ++j) {
      int idx = j * 256 + t;
      int rr = idx >> 3;          // 8 chunks per 64-elem row
      int c8 = (idx & 7) * 8;
      async_b128(&As[buf][rr][c8], &A[(long long)(m0 + rr) * lda + k0 + c8]);
    }
    int nn = t >> 3;
    int c8b = (t & 7) * 8;
    async_b128(&Bs[buf][nn][c8b], &Bt[(long long)(n0 + nn) * ldbt + k0 + c8b]);
  };

  v8f acc = {};
  stageAB(0, 0);
  wait_async();
  __syncthreads();
  int cur = 0;
  for (int k0 = 0; k0 < K; k0 += 64) {
    if (k0 + 64 < K) stageAB(cur ^ 1, k0 + 64);  // prefetch next tile (async)
#pragma unroll
    for (int s = 0; s < 2; ++s) {
      const int kb = s * 32;
      v16bf af = cat16(*(const v8bf*)&As[cur][ar][kb + g * 8],
                       *(const v8bf*)&As[cur][ar][kb + 16 + g * 8]);
      v16bf bf_ = cat16(*(const v8bf*)&Bs[cur][bn][kb + g * 8],
                        *(const v8bf*)&Bs[cur][bn][kb + 16 + g * 8]);
      acc = wmma_bf16(af, bf_, acc);
    }
    wait_async();
    __syncthreads();
    cur ^= 1;
  }

  // epilogue (C layout: elem e -> M = rowTile*16 + g*8 + e, N = colTile*16 + r)
  const int ncol = n0 + colTile * 16 + r;
  const int mBase = m0 + rowTile * 16 + g * 8;
  if constexpr (EPI == 0) {
    const long long batch = m0 / rowsPerBatch;
    const int rowBase = (m0 % rowsPerBatch) + rowTile * 16 + g * 8;
    const long long cbase =
        batch * batchStrideC + (long long)ncol * colStrideC;
#pragma unroll
    for (int e = 0; e < 8; ++e) {
      Cb[cbase + (long long)(rowBase + e) * rowStrideC] = (__bf16)acc[e];
    }
  } else if constexpr (EPI == 1) {
    const float bsum = bias[ncol];
#pragma unroll
    for (int e = 0; e < 8; ++e) {
      long long ofs = (long long)(mBase + e) * N + ncol;
      Cf[ofs] = acc[e] + bsum + residual[ofs];
    }
  } else {  // EPI == 2
    const float bsum = bias[ncol];
#pragma unroll
    for (int e = 0; e < 8; ++e) {
      float vv = acc[e] + bsum;
      vv = vv > 0.f ? vv : 0.f;
      Cb[(long long)(mBase + e) * N + ncol] = (__bf16)vv;
    }
  }
}

// ---------------- flash-attention (streaming softmax, double-buffered K/V) ----------------
// q,k: bf16 [B,H,S,DK]; vt: bf16 [B,H,DK,S] (pre-transposed by V-projection GEMM).
// o: bf16 [B,S,D] (heads concatenated).
// Block = 256 threads; wave w handles query rows [blockIdx.x*128 + w*16, +16).
__global__ __launch_bounds__(256) void attn_kernel(
    const __bf16* __restrict__ q, const __bf16* __restrict__ k,
    const __bf16* __restrict__ vt, const unsigned char* __restrict__ mask,
    __bf16* __restrict__ o) {
  __shared__ __bf16 Ks[2][32][40];    // [buf][key][kdim]
  __shared__ __bf16 Vt[2][32][40];    // [buf][vcol][key]
  __shared__ __bf16 Ps[8][16][40];    // [wave][qrow][key] (wave-private)
  const int t = threadIdx.x;
  const int lane = t & 31;
  const int w = t >> 5;
  const int g = lane >> 4;
  const int r = lane & 15;
  const int b = blockIdx.z;
  const int h = blockIdx.y;
  const int qs0 = (blockIdx.x * 8 + w) * 16;
  const long long bhBase = ((long long)(b * H_ + h)) * S_ * DK_;   // q/k layout
  const long long vtBase = ((long long)(b * H_ + h)) * DK_ * S_;   // vt layout
  const long long mBase = (long long)b * S_ * S_;
  const float scale = 0.17677669529663687f;  // 1/sqrt(DK)

  auto stageKV = [&](int buf, int kt) {
    if (t < 128) {
      int rr = t >> 2;          // key 0..31
      int c8 = (t & 3) * 8;     // kdim
      async_b128(&Ks[buf][rr][c8], &k[bhBase + (long long)(kt + rr) * DK_ + c8]);
    } else {
      int idx = t - 128;
      int vc = idx >> 2;        // vcol 0..31
      int c8 = (idx & 3) * 8;   // key
      async_b128(&Vt[buf][vc][c8], &vt[vtBase + (long long)vc * S_ + kt + c8]);
    }
  };

  // Q fragment (A layout) straight from bf16 global: 2x b128
  v16bf aq;
  {
    const __bf16* qrow = q + bhBase + (long long)(qs0 + r) * DK_;
    aq = cat16(*(const v8bf*)&qrow[g * 8], *(const v8bf*)&qrow[16 + g * 8]);
  }

  float rowmax[8], rowsum[8];
  v8f olo = {}, ohi = {};
#pragma unroll
  for (int e = 0; e < 8; ++e) { rowmax[e] = NEG_INF; rowsum[e] = 0.f; }

  stageKV(0, 0);
  wait_async();
  __syncthreads();
  int cur = 0;
  for (int kt = 0; kt < S_; kt += 32) {
    if (kt + 32 < S_) stageKV(cur ^ 1, kt + 32);  // prefetch next K/V tiles (async)

    // scores for 32 keys: two 16x16 WMMAs (B[kdim][n] = K[kt+n][kdim])
    v16bf b0 = cat16(*(const v8bf*)&Ks[cur][r][g * 8],
                     *(const v8bf*)&Ks[cur][r][16 + g * 8]);
    v16bf b1 = cat16(*(const v8bf*)&Ks[cur][16 + r][g * 8],
                     *(const v8bf*)&Ks[cur][16 + r][16 + g * 8]);
    v8f s0 = {}, s1 = {};
    s0 = wmma_bf16(aq, b0, s0);
    s1 = wmma_bf16(aq, b1, s1);

    // online softmax per row (C layout: row = g*8+e, col = lane&15)
#pragma unroll
    for (int e = 0; e < 8; ++e) {
      int sq = qs0 + g * 8 + e;
      float x0 = s0[e] * scale;
      float x1 = s1[e] * scale;
      if (mask[mBase + (long long)sq * S_ + (kt + r)]) x0 = NEG_INF;
      if (mask[mBase + (long long)sq * S_ + (kt + 16 + r)]) x1 = NEG_INF;
      float mx = fmaxf(x0, x1);
      mx = fmaxf(mx, __shfl_xor(mx, 1, 32));
      mx = fmaxf(mx, __shfl_xor(mx, 2, 32));
      mx = fmaxf(mx, __shfl_xor(mx, 4, 32));
      mx = fmaxf(mx, __shfl_xor(mx, 8, 32));
      float mnew = fmaxf(rowmax[e], mx);
      float corr = __expf(rowmax[e] - mnew);
      float p0 = __expf(x0 - mnew);
      float p1 = __expf(x1 - mnew);
      float ps = p0 + p1;
      ps += __shfl_xor(ps, 1, 32);
      ps += __shfl_xor(ps, 2, 32);
      ps += __shfl_xor(ps, 4, 32);
      ps += __shfl_xor(ps, 8, 32);
      rowsum[e] = rowsum[e] * corr + ps;
      rowmax[e] = mnew;
      olo[e] *= corr;
      ohi[e] *= corr;
      Ps[w][g * 8 + e][r] = (__bf16)p0;
      Ps[w][g * 8 + e][16 + r] = (__bf16)p1;
    }
    lds_fence();  // wave-private Ps transpose: DS in-order per wave, stop reordering

    // O += P(16x32) * V(32x32): A from Ps, B from Vt
    v16bf ap = cat16(*(const v8bf*)&Ps[w][r][g * 8],
                     *(const v8bf*)&Ps[w][r][16 + g * 8]);
    v16bf bl = cat16(*(const v8bf*)&Vt[cur][r][g * 8],
                     *(const v8bf*)&Vt[cur][r][16 + g * 8]);
    v16bf bh = cat16(*(const v8bf*)&Vt[cur][16 + r][g * 8],
                     *(const v8bf*)&Vt[cur][16 + r][16 + g * 8]);
    olo = wmma_bf16(ap, bl, olo);
    ohi = wmma_bf16(ap, bh, ohi);

    wait_async();
    __syncthreads();  // all waves done with cur; next buffer fully staged
    cur ^= 1;
  }

  // write o[b, sq, h*DK + n] (bf16)
#pragma unroll
  for (int e = 0; e < 8; ++e) {
    int sq = qs0 + g * 8 + e;
    float inv = 1.0f / rowsum[e];
    long long obase = ((long long)b * S_ + sq) * D_ + h * DK_;
    o[obase + r] = (__bf16)(olo[e] * inv);
    o[obase + 16 + r] = (__bf16)(ohi[e] * inv);
  }
}

// ---------------- row LayerNorm over D=256 (fp32 + bf16 outputs) ----------------
__global__ __launch_bounds__(256) void ln_kernel(const float* __restrict__ in,
                                                 const float* __restrict__ gamma,
                                                 const float* __restrict__ beta,
                                                 float* __restrict__ outf,
                                                 __bf16* __restrict__ outb) {
  __shared__ float red[256];
  const int row = blockIdx.x;
  const int t = threadIdx.x;
  float val = in[(long long)row * D_ + t];
  red[t] = val;
  __syncthreads();
#pragma unroll
  for (int s = 128; s > 0; s >>= 1) {
    if (t < s) red[t] += red[t + s];
    __syncthreads();
  }
  float mu = red[0] * (1.0f / D_);
  __syncthreads();
  float d = val - mu;
  red[t] = d * d;
  __syncthreads();
#pragma unroll
  for (int s = 128; s > 0; s >>= 1) {
    if (t < s) red[t] += red[t + s];
    __syncthreads();
  }
  float var = red[0] * (1.0f / D_);
  float y = d * rsqrtf(var + 1e-7f) * gamma[t] + beta[t];
  outf[(long long)row * D_ + t] = y;
  outb[(long long)row * D_ + t] = (__bf16)y;
}

// ---------------- orchestration ----------------
extern "C" void kernel_launch(void* const* d_in, const int* in_sizes, int n_in,
                              void* d_out, int out_size, void* d_ws, size_t ws_size,
                              hipStream_t stream) {
  (void)in_sizes; (void)n_in; (void)out_size; (void)ws_size;
  const float* x_in = (const float*)d_in[0];
  const unsigned char* mask = (const unsigned char*)d_in[1];
  const float* pe = (const float*)d_in[2];
  const float* Wq = (const float*)d_in[3];
  const float* Wk = (const float*)d_in[4];
  const float* Wv = (const float*)d_in[5];
  const float* Wo = (const float*)d_in[6];
  const float* bo = (const float*)d_in[7];
  const float* ln1g = (const float*)d_in[8];
  const float* ln1b = (const float*)d_in[9];
  const float* W1 = (const float*)d_in[10];
  const float* b1 = (const float*)d_in[11];
  const float* W2 = (const float*)d_in[12];
  const float* b2 = (const float*)d_in[13];
  const float* ln2g = (const float*)d_in[14];
  const float* ln2b = (const float*)d_in[15];
  float* out = (float*)d_out;

  const long long M = (long long)B_ * S_;  // 4096
  // workspace layout (all chunks 16B-aligned)
  char* p = (char*)d_ws;
  float* xb = (float*)p;  p += M * D_ * 4;            // fp32 activations (residual/LN)
  float* tb = (float*)p;  p += M * D_ * 4;            // fp32 pre-LN temp
  __bf16* xbh = (__bf16*)p; p += M * D_ * 2;          // bf16 activations (GEMM A)
  __bf16* obh = (__bf16*)p; p += M * D_ * 2;          // bf16 attention out
  __bf16* qbh = (__bf16*)p; p += M * D_ * 2;          // [B,H,S,DK]
  __bf16* kbh = (__bf16*)p; p += M * D_ * 2;          // [B,H,S,DK]
  __bf16* vtbh = (__bf16*)p; p += M * D_ * 2;         // [B,H,DK,S]
  __bf16* hbh = (__bf16*)p; p += M * F_ * 2;          // bf16 FFN hidden
  __bf16* wqt = (__bf16*)p; p += (long long)L_ * H_ * D_ * DK_ * 2;  // [L,H,DK,D]
  __bf16* wkt = (__bf16*)p; p += (long long)L_ * H_ * D_ * DK_ * 2;
  __bf16* wvt = (__bf16*)p; p += (long long)L_ * H_ * D_ * DK_ * 2;
  __bf16* wot = (__bf16*)p; p += (long long)L_ * D_ * D_ * 2;        // [L,D,D] (n,k)
  __bf16* w1t = (__bf16*)p; p += (long long)L_ * D_ * F_ * 2;        // [L,F,D]
  __bf16* w2t = (__bf16*)p; p += (long long)L_ * F_ * D_ * 2;        // [L,D,F]

  // weight convert + transpose (bf16, [N][K])
  transpose_bf16_kernel<<<dim3((D_ * DK_ + 255) / 256, L_ * H_), 256, 0, stream>>>(Wq, wqt, D_, DK_);
  transpose_bf16_kernel<<<dim3((D_ * DK_ + 255) / 256, L_ * H_), 256, 0, stream>>>(Wk, wkt, D_, DK_);
  transpose_bf16_kernel<<<dim3((D_ * DK_ + 255) / 256, L_ * H_), 256, 0, stream>>>(Wv, wvt, D_, DK_);
  transpose_bf16_kernel<<<dim3((D_ * D_ + 255) / 256, L_), 256, 0, stream>>>(Wo, wot, D_, D_);
  transpose_bf16_kernel<<<dim3((D_ * F_ + 255) / 256, L_), 256, 0, stream>>>(W1, w1t, D_, F_);
  transpose_bf16_kernel<<<dim3((F_ * D_ + 255) / 256, L_), 256, 0, stream>>>(W2, w2t, F_, D_);

  addpe_kernel<<<dim3((unsigned)(M * D_ / 256)), 256, 0, stream>>>(x_in, pe, xb, xbh);

  for (int l = 0; l < L_; ++l) {
    const __bf16* wqt_l = wqt + (long long)l * H_ * DK_ * D_;
    const __bf16* wkt_l = wkt + (long long)l * H_ * DK_ * D_;
    const __bf16* wvt_l = wvt + (long long)l * H_ * DK_ * D_;
    const __bf16* wot_l = wot + (long long)l * D_ * D_;
    const __bf16* w1t_l = w1t + (long long)l * F_ * D_;
    const __bf16* w2t_l = w2t + (long long)l * D_ * F_;
    const float* bo_l = bo + (long long)l * D_;
    const float* b1_l = b1 + (long long)l * F_;
    const float* b2_l = b2 + (long long)l * D_;

    // Q/K projections -> bf16 [B,H,S,DK]; V -> bf16 transposed [B,H,DK,S]
    for (int h = 0; h < H_; ++h) {
      gemm_wmma_kernel<0><<<dim3(1, (unsigned)(M / 64)), 256, 0, stream>>>(
          xbh, D_, wqt_l + (long long)h * DK_ * D_, D_, nullptr, nullptr,
          nullptr, qbh + (long long)h * S_ * DK_,
          S_, (long long)H_ * S_ * DK_, DK_, 1, DK_, D_);
      gemm_wmma_kernel<0><<<dim3(1, (unsigned)(M / 64)), 256, 0, stream>>>(
          xbh, D_, wkt_l + (long long)h * DK_ * D_, D_, nullptr, nullptr,
          nullptr, kbh + (long long)h * S_ * DK_,
          S_, (long long)H_ * S_ * DK_, DK_, 1, DK_, D_);
      gemm_wmma_kernel<0><<<dim3(1, (unsigned)(M / 64)), 256, 0, stream>>>(
          xbh, D_, wvt_l + (long long)h * DK_ * D_, D_, nullptr, nullptr,
          nullptr, vtbh + (long long)h * DK_ * S_,
          S_, (long long)H_ * DK_ * S_, 1, S_, DK_, D_);
    }

    // attention -> obh bf16 [B,S,D]
    attn_kernel<<<dim3(S_ / 128, H_, B_), 256, 0, stream>>>(qbh, kbh, vtbh, mask, obh);

    // out-proj + bias + residual -> tb (fp32); then LN1 -> xb + xbh
    gemm_wmma_kernel<1><<<dim3(D_ / 32, (unsigned)(M / 64)), 256, 0, stream>>>(
        obh, D_, wot_l, D_, bo_l, xb, tb, nullptr, 0, 0, 0, 0, D_, D_);
    ln_kernel<<<dim3((unsigned)M), 256, 0, stream>>>(tb, ln1g + (long long)l * D_,
                                                     ln1b + (long long)l * D_, xb, xbh);

    // FFN: relu(x W1 + b1) -> hbh (bf16); hbh W2 + b2 + x -> tb; LN2
    gemm_wmma_kernel<2><<<dim3(F_ / 32, (unsigned)(M / 64)), 256, 0, stream>>>(
        xbh, D_, w1t_l, D_, b1_l, nullptr, nullptr, hbh, 0, 0, 0, 0, F_, D_);
    gemm_wmma_kernel<1><<<dim3(D_ / 32, (unsigned)(M / 64)), 256, 0, stream>>>(
        hbh, F_, w2t_l, F_, b2_l, xb, tb, nullptr, 0, 0, 0, 0, D_, F_);
    float* lnOut = (l == L_ - 1) ? out : xb;
    ln_kernel<<<dim3((unsigned)M), 256, 0, stream>>>(tb, ln2g + (long long)l * D_,
                                                     ln2b + (long long)l * D_, lnOut, xbh);
  }
}
